// sqgkt_17832704213617
// MI455X (gfx1250) — matrix operand order repaired and checked by reading, not compiled, not run on hardware
//
#include <hip/hip_runtime.h>
#include <hip/hip_bf16.h>

typedef __attribute__((ext_vector_type(16))) _Float16 v16h;
typedef __attribute__((ext_vector_type(8)))  float    v8f;

static constexpr int Bv   = 64;
static constexpr int Sv   = 100;
static constexpr int Tv   = 99;    // S-1 steps
static constexpr int NQv  = 2000;
static constexpr int NSKv = 100;
static constexpr int Dv   = 100;
static constexpr int PADv = 112;   // padded M/N (7 tiles of 16)
static constexpr int KPv  = 128;   // padded K   (4 steps of 32)

// ---------------------------------------------------------------- reductions
__device__ __forceinline__ float block_sum128(float* red, float v) {
    int tid = threadIdx.x;
    red[tid] = v; __syncthreads();
    for (int s = 64; s > 0; s >>= 1) {
        if (tid < s) red[tid] += red[tid + s];
        __syncthreads();
    }
    float r = red[0]; __syncthreads();
    return r;
}
__device__ __forceinline__ float block_max128(float* red, float v) {
    int tid = threadIdx.x;
    red[tid] = v; __syncthreads();
    for (int s = 64; s > 0; s >>= 1) {
        if (tid < s) red[tid] = fmaxf(red[tid], red[tid + s]);
        __syncthreads();
    }
    float r = red[0]; __syncthreads();
    return r;
}

__device__ __forceinline__ float fast_sigmoid(float x) {
    return __builtin_amdgcn_rcpf(1.f + __expf(-x));   // v_rcp_f32, no IEEE divide
}

// ---------------------------------------------------------------- zero fill (16B stores)
__global__ void zero16_kernel(uint4* p, long n16) {
    long i = (long)blockIdx.x * blockDim.x + threadIdx.x;
    long stride = (long)gridDim.x * blockDim.x;
    uint4 z = {0u, 0u, 0u, 0u};
    for (; i < n16; i += stride) p[i] = z;
}

// ---------------------------------------------------------------- skill bitmasks
__global__ void qbits_kernel(const float* __restrict__ qs_table, unsigned* __restrict__ qbits) {
    int q = blockIdx.x * blockDim.x + threadIdx.x;
    if (q >= NQv) return;
    for (int w = 0; w < 4; w++) {
        unsigned bits = 0;
        for (int k = 0; k < 32; k++) {
            int kk = w * 32 + k;
            if (kk < NSKv && qs_table[q * NSKv + kk] > 0.f) bits |= (1u << k);
        }
        qbits[q * 4 + w] = bits;
    }
}

// ---------------------------------------------------------------- weight f16 pre-pack
// W16 layout: [4][112][128] f16; m=0..2 -> agg_W[m] (row d, col k), m=3 -> aggL_W
__global__ void wprep_kernel(const float* __restrict__ aggW, const float* __restrict__ aggLW,
                             _Float16* __restrict__ W16) {
    int idx = blockIdx.x * blockDim.x + threadIdx.x;
    const int total = 4 * PADv * KPv;
    if (idx >= total) return;
    int m   = idx / (PADv * KPv);
    int rem = idx % (PADv * KPv);
    int d = rem >> 7, k = rem & 127;
    float v = 0.f;
    if (d < Dv && k < Dv) v = (m < 3) ? aggW[(m * Dv + d) * Dv + k] : aggLW[d * Dv + k];
    W16[idx] = (_Float16)v;
}

// ---------------------------------------------------------------- constant precompute
__global__ __launch_bounds__(256) void const_prep_kernel(
    const float* __restrict__ emb_s, const float* __restrict__ qW, const float* __restrict__ qb,
    const float* __restrict__ kW, const float* __restrict__ kb, const float* __restrict__ wW,
    float* __restrict__ qwa, float* __restrict__ kwa, float* __restrict__ cst,
    float* __restrict__ lqs, _Float16* __restrict__ embs16)
{
    int tid = threadIdx.x;
    if (tid < 128) {
        float s = 0.f;
        if (tid < Dv) for (int d = 0; d < Dv; d++) s += wW[d] * qW[d * Dv + tid];
        qwa[tid] = s;
    } else {
        int e = tid - 128;
        float s = 0.f;
        if (e < Dv) for (int d = 0; d < Dv; d++) s += wW[Dv + d] * kW[d * Dv + e];
        kwa[e] = s;
    }
    if (tid == 0) { float s = 0.f; for (int d = 0; d < Dv; d++) s += wW[d] * qb[d];      cst[0] = s; }
    if (tid == 1) { float s = 0.f; for (int d = 0; d < Dv; d++) s += wW[Dv + d] * kb[d]; cst[1] = s; }
    __syncthreads();
    if (tid < PADv) {
        float s = 0.f;
        if (tid < NSKv) { s = cst[0]; for (int e = 0; e < Dv; e++) s += emb_s[tid * Dv + e] * qwa[e]; }
        lqs[tid] = s;
    }
    for (int idx = tid; idx < PADv * KPv; idx += 256) {
        int r = idx >> 7, d = idx & 127;
        float v = (r < NSKv && d < Dv) ? emb_s[r * Dv + d] : 0.f;
        embs16[idx] = (_Float16)v;
    }
}

// ---------------------------------------------------------------- WMMA pass inside phase 1
// otile[r][d] = sum_k atile[r][k] * W16[d][k]; atile LDS [16][128] f16, otile LDS [16][112] f32.
// 4 waves split the 7 N-tiles; epilogue is 8 unconditional ds_store_b32 (no divergence).
__device__ __forceinline__ void gemm_pass(
    const _Float16* atile, const _Float16* __restrict__ W16,
    float* otile, int lane, int wave)
{
    int half = lane >> 4, l15 = lane & 15;
    for (int tt = wave; tt < 7; tt += 4) {
        v8f acc = {};
        #pragma unroll
        for (int kk = 0; kk < 4; kk++) {
            int k0 = kk * 32;
            v16h a, bf;
            // A (16x32 f16): lane<16 -> M=l15, K = k0+0..7 / k0+16..23 ; lane>=16 -> K = +8 / +24
            const _Float16* ap = atile + l15 * KPv + k0 + half * 8;
            #pragma unroll
            for (int i = 0; i < 8; i++) a[i]     = ap[i];
            #pragma unroll
            for (int i = 0; i < 8; i++) a[8 + i] = ap[16 + i];
            // B (32x16 f16): lane n holds column d=tt*16+l15, K = k0+0..7 / k0+16..23 ; lane>=16 -> K = +8 / +24
            const _Float16* bp = W16 + (tt * 16 + l15) * KPv + k0 + half * 8;
            #pragma unroll
            for (int i = 0; i < 8; i++) bf[i]     = bp[i];
            #pragma unroll
            for (int i = 0; i < 8; i++) bf[8 + i] = bp[16 + i];
            acc = __builtin_amdgcn_wmma_f32_16x16x32_f16(false, a, false, bf,
                                                         (short)0, acc, false, false);
        }
        // C/D layout: VGPR i -> M = half*8+i, N = l15 ; store raw accumulator, padded
        float* op = otile + (half * 8) * PADv + tt * 16 + l15;
        #pragma unroll
        for (int i = 0; i < 8; i++) op[i * PADv] = acc[i];
    }
}

// ---------------------------------------------------------------- GNN (device, fully inlined)
// Triangular GIKT aggregation, 3 hops, fan-out 4. T3 == T1 by structure.
__device__ __forceinline__ void run_gnn(
    const float* __restrict__ EA, const float* __restrict__ EB,
    const int* __restrict__ T1, const int* __restrict__ T2,
    int root,
    const _Float16* __restrict__ Wagg16,  // [3][112*128]
    const float* __restrict__ aggb,       // [3][100]
    const _Float16* __restrict__ WaggL16, // [112*128]
    const float* __restrict__ aggLb,
    bool useW, const float* wuq,
    float* L0, float* L1, float* L2, float* L3,
    int* I1, int* I2, int* I3,
    _Float16* atile, float* otile, float* outv)
{
    int tid = threadIdx.x, lane = tid & 31, wave = tid >> 5;
    if (tid < 4) I1[tid] = T1[root * 4 + tid];
    __syncthreads();
    if (tid < 16) I2[tid] = T2[I1[tid >> 2] * 4 + (tid & 3)];
    __syncthreads();
    if (tid < 64) I3[tid] = T1[I2[tid >> 2] * 4 + (tid & 3)];
    __syncthreads();
    for (int i = tid; i < 100;  i += 128) L0[i] = EA[root * Dv + i];
    for (int i = tid; i < 400;  i += 128) L1[i] = EB[I1[i / 100] * Dv + (i % 100)];
    for (int i = tid; i < 1600; i += 128) L2[i] = EA[I2[i / 100] * Dv + (i % 100)];
    for (int i = tid; i < 6400; i += 128) L3[i] = EB[I3[i / 100] * Dv + (i % 100)];
    __syncthreads();
    float* lvl[4] = {L0, L1, L2, L3};
    const int rowsArr[3] = {1, 4, 16};
    for (int i = 0; i < 3; i++) {
        for (int j = 0; j < 3 - i; j++) {
            float* P = lvl[j];
            float* C = lvl[j + 1];
            int rows = rowsArr[j];
            // stage (weighted mean of children + parent) into f16 A-tile, zero padded
            for (int idx = tid; idx < 16 * KPv; idx += 128) {
                int r = idx >> 7, k = idx & 127;
                float v = 0.f;
                if (r < rows && k < Dv) {
                    float s = 0.f;
                    #pragma unroll
                    for (int c = 0; c < 4; c++) {
                        float cv = C[(r * 4 + c) * Dv + k];
                        s += useW ? (wuq[c] * cv) : cv;
                    }
                    v = s * 0.25f + P[r * Dv + k];
                }
                atile[idx] = (_Float16)v;
            }
            __syncthreads();
            gemm_pass(atile, Wagg16 + j * (PADv * KPv), otile, lane, wave);
            __syncthreads();
            // bias + tanh on the valid region only (uniform bounds, no epilogue divergence)
            const float* bb = aggb + j * Dv;
            for (int idx = tid; idx < rows * Dv; idx += 128) {
                int r = idx / Dv, d = idx - r * Dv;
                P[idx] = tanhf(otile[r * PADv + d] + bb[d]);
            }
            __syncthreads();
        }
    }
    // final linear: rows = 1
    for (int idx = tid; idx < 16 * KPv; idx += 128) {
        int r = idx >> 7, k = idx & 127;
        atile[idx] = (_Float16)((r == 0 && k < Dv) ? L0[k] : 0.f);
    }
    __syncthreads();
    gemm_pass(atile, WaggL16, otile, lane, wave);
    __syncthreads();
    if (tid < Dv) outv[tid] = tanhf(otile[tid] + aggLb[tid]);
    __syncthreads();
}

// ---------------------------------------------------------------- phase 1
// One block (4 waves) per (b,t): GNNs (WMMA linears) -> e1,e2 -> LSTM(h0=0) -> h
__global__ __launch_bounds__(128) void phase1_kernel(
    const float* __restrict__ emb_q, const float* __restrict__ emb_q2,
    const float* __restrict__ emb_s, const float* __restrict__ emb_u,
    const float* __restrict__ emb_r,
    const float* __restrict__ w1p, const float* __restrict__ w2p,
    const float* __restrict__ wcp, const float* __restrict__ wpp, const float* __restrict__ wnp,
    const float* __restrict__ Wih, const float* __restrict__ bih, const float* __restrict__ bhh,
    const _Float16* __restrict__ W16,          // [4][112*128]: agg_W[0..2], aggL_W
    const float* __restrict__ aggb, const float* __restrict__ aggLb,
    const float* __restrict__ uq_table,
    const int* __restrict__ q_nb, const int* __restrict__ s_nb,
    const int* __restrict__ u_nb, const int* __restrict__ q_nb2,
    const int* __restrict__ user, const int* __restrict__ question,
    const int* __restrict__ response, const int* __restrict__ mask,
    const float* __restrict__ qwa, const float* __restrict__ kwa, const float* __restrict__ cst,
    _Float16* __restrict__ h16, _Float16* __restrict__ eq16,
    float* __restrict__ lkv, float* __restrict__ lq0)
{
    __shared__ float L0[100], L1[400], L2[1600], L3[6400];
    __shared__ int I1[4], I2[16], I3[64];
    __shared__ _Float16 atile[16 * KPv];
    __shared__ float otile[16 * PADv];
    __shared__ float gq[100], gu[100], xv[200];
    __shared__ float wuq[4];
    __shared__ float red[128];

    int t = blockIdx.x, b = blockIdx.y, tid = threadIdx.x;
    if (tid < 4) {
        wuq[tid] = wcp[0] * uq_table[(b * 4 + tid) * 3 + 0]
                 + wpp[0] * uq_table[(b * 4 + tid) * 3 + 1]
                 + wnp[0] * uq_table[(b * 4 + tid) * 3 + 2];
    }
    int q  = question[b * Sv + t];
    int u  = user[b * Sv + t];
    int rr = response[b * Sv + t];
    int mk = mask[b * Sv + t];
    __syncthreads();

    const _Float16* Wagg16  = W16;
    const _Float16* WaggL16 = W16 + 3 * (PADv * KPv);
    run_gnn(emb_q, emb_s, q_nb, s_nb, q, Wagg16, aggb, WaggL16, aggLb, false, wuq,
            L0, L1, L2, L3, I1, I2, I3, atile, otile, gq);
    run_gnn(emb_u, emb_q2, u_nb, q_nb2, u, Wagg16, aggb, WaggL16, aggLb, true, wuq,
            L0, L1, L2, L3, I1, I2, I3, atile, otile, gu);

    float w1 = w1p[0], w2 = w2p[0];
    if (tid < Dv) {
        float e1 = (mk == 1) ? gq[tid] : emb_q[q * Dv + tid];
        float e2 = (mk == 1) ? gu[tid] : emb_q2[q * Dv + tid];
        xv[tid]      = w1 * e1 + w2 * e2;
        xv[Dv + tid] = emb_r[rr * Dv + tid];
    }
    __syncthreads();

    float hval = 0.f;
    if (tid < Dv) {
        float gi = bih[tid]       + bhh[tid];
        float gg = bih[200 + tid] + bhh[200 + tid];
        float go = bih[300 + tid] + bhh[300 + tid];
        const float* wi = Wih + (long)tid * 200;
        const float* wg = Wih + (long)(200 + tid) * 200;
        const float* wo = Wih + (long)(300 + tid) * 200;
        for (int k = 0; k < 200; k++) {
            float xk = xv[k];
            gi += xk * wi[k]; gg += xk * wg[k]; go += xk * wo[k];
        }
        float si = fast_sigmoid(gi);
        float tg = tanhf(gg);
        float so = fast_sigmoid(go);
        hval = so * tanhf(si * tg);
        h16[((long)(b * PADv) + t) * KPv + tid] = (_Float16)hval;
    }
    float sk = block_sum128(red, (tid < Dv) ? hval * kwa[tid] : 0.f);
    if (tid == 0) lkv[b * PADv + t] = sk + cst[1];

    int qn = question[b * Sv + t + 1];
    float eqd = 0.f;
    if (tid < Dv) {
        eqd = emb_q[qn * Dv + tid];
        eq16[((long)(b * PADv) + t) * KPv + tid] = (_Float16)eqd;
    }
    float sq = block_sum128(red, (tid < Dv) ? eqd * qwa[tid] : 0.f);
    if (tid == 0) lq0[b * PADv + t] = sq + cst[0];
}

// ---------------------------------------------------------------- WMMA batched GEMM + sigmoid
// C[b][m][n] = sigmoid( sum_d A[b][m][d] * Bm[b][n][d] ), A/Bm padded f16 [PAD][128]
__global__ __launch_bounds__(32) void wmma_sig_kernel(
    const _Float16* __restrict__ A, long strideA,
    const _Float16* __restrict__ Bm,
    float* __restrict__ C)
{
    int mt = blockIdx.x, nt = blockIdx.y, b = blockIdx.z;
    const _Float16* Ab = A  + (long)b * strideA;
    const _Float16* Bb = Bm + (long)b * (PADv * KPv);
    int lane = threadIdx.x;
    int half = lane >> 4;
    int l15  = lane & 15;
    int arow = mt * 16 + l15; // M
    int brow = nt * 16 + l15; // N  (column of C == h16 row)
    v8f acc = {};
    for (int k0 = 0; k0 < KPv; k0 += 32) {
        v16h a, bf;
        const _Float16* ap = Ab + (long)arow * KPv + k0 + half * 8;
        #pragma unroll
        for (int i = 0; i < 8; i++) a[i]     = ap[i];
        #pragma unroll
        for (int i = 0; i < 8; i++) a[8 + i] = ap[16 + i];
        const _Float16* bp = Bb + (long)brow * KPv + k0 + half * 8;
        #pragma unroll
        for (int i = 0; i < 8; i++) bf[i]     = bp[i];
        #pragma unroll
        for (int i = 0; i < 8; i++) bf[8 + i] = bp[16 + i];
        acc = __builtin_amdgcn_wmma_f32_16x16x32_f16(false, a, false, bf,
                                                     (short)0, acc, false, false);
    }
    float* Cb = C + ((long)b * PADv + mt * 16) * PADv + nt * 16;
    #pragma unroll
    for (int i = 0; i < 8; i++) {
        int m = half * 8 + i;
        Cb[(long)m * PADv + l15] = fast_sigmoid(acc[i]);
    }
}

// ---------------------------------------------------------------- finalize
// Rank-1-factorized masked softmax:  y = (sum_ij wq_i wk_j sig_ij) / (sum wq)(sum wk)
__global__ __launch_bounds__(128) void finalize_kernel(
    const int* __restrict__ question,
    const unsigned* __restrict__ qbits,
    const float* __restrict__ lkv, const float* __restrict__ lq0,
    const float* __restrict__ lqs,
    const float* __restrict__ Sg, const float* __restrict__ Q0g,
    float* __restrict__ out)
{
    __shared__ float red[128];
    __shared__ float wqList[104];
    __shared__ int   kList[104];
    __shared__ int   nvi_s;
    __shared__ float sumWq_s;
    __shared__ unsigned qb[4];

    int t = blockIdx.x, b = blockIdx.y, tid = threadIdx.x;
    int qn = question[b * Sv + t + 1];
    if (tid < 4) qb[tid] = qbits[qn * 4 + tid];
    __syncthreads();

    int ncols = (t == 0) ? 1 : t;
    float lk = -1e30f;
    int timej = 0;
    bool validc = false;
    if (tid < ncols) {
        timej = (tid == 0) ? t : tid;
        if (tid == 0) validc = true;
        else {
            int qj = question[b * Sv + tid];
            unsigned o = 0;
            for (int w = 0; w < 4; w++) o |= qb[w] & qbits[qj * 4 + w];
            validc = (o != 0u);
        }
        if (validc) lk = lkv[b * PADv + timej];
    }
    float mlk   = block_max128(red, lk);
    float wk    = validc ? __expf(lk - mlk) : 0.f;
    float sumWk = block_sum128(red, wk);

    if (tid == 0) {
        float m = lq0[b * PADv + t];
        for (int k = 0; k < NSKv; k++)
            if (qb[k >> 5] & (1u << (k & 31))) m = fmaxf(m, lqs[k]);
        float s = 0.f; int n = 0;
        wqList[0] = __expf(lq0[b * PADv + t] - m); kList[0] = -1; s = wqList[0]; n = 1;
        for (int k = 0; k < NSKv; k++)
            if (qb[k >> 5] & (1u << (k & 31))) {
                wqList[n] = __expf(lqs[k] - m); kList[n] = k; s += wqList[n]; n++;
            }
        nvi_s = n; sumWq_s = s;
    }
    __syncthreads();

    float part = 0.f;
    if (tid < ncols && validc) {
        const float* Qrow = Q0g + ((long)b * PADv + t) * PADv;
        int nvi = nvi_s;
        for (int n = 0; n < nvi; n++) {
            int k = kList[n];
            float sig = (k < 0) ? Qrow[timej]
                                : Sg[((long)b * PADv + k) * PADv + timej];
            part += wqList[n] * sig;
        }
        part *= wk;
    }
    float num = block_sum128(red, part);
    if (tid == 0) {
        out[b * Sv + t + 1] = num / (sumWq_s * sumWk);
        if (t == 0) out[b * Sv] = 0.5f;
    }
}

// ---------------------------------------------------------------- launch
extern "C" void kernel_launch(void* const* d_in, const int* in_sizes, int n_in,
                              void* d_out, int out_size, void* d_ws, size_t ws_size,
                              hipStream_t stream) {
    const float* emb_q   = (const float*)d_in[0];
    const float* emb_q2  = (const float*)d_in[1];
    const float* emb_s   = (const float*)d_in[2];
    const float* emb_u   = (const float*)d_in[3];
    const float* emb_r   = (const float*)d_in[4];
    const float* w1p     = (const float*)d_in[5];
    const float* w2p     = (const float*)d_in[6];
    const float* wcp     = (const float*)d_in[7];
    const float* wpp     = (const float*)d_in[8];
    const float* wnp     = (const float*)d_in[9];
    const float* Wih     = (const float*)d_in[10];
    // d_in[11] = Whh (numerically unused, h0 = 0)
    const float* bih     = (const float*)d_in[12];
    const float* bhh     = (const float*)d_in[13];
    const float* aggW    = (const float*)d_in[14];
    const float* aggb    = (const float*)d_in[15];
    const float* aggLW   = (const float*)d_in[16];
    const float* aggLb   = (const float*)d_in[17];
    const float* qW      = (const float*)d_in[18];
    const float* qb      = (const float*)d_in[19];
    const float* kW      = (const float*)d_in[20];
    const float* kb      = (const float*)d_in[21];
    const float* wW      = (const float*)d_in[22];
    // d_in[23] = wb (constant offset, cancels in softmax)
    const float* uq_tab  = (const float*)d_in[24];
    const float* qs_tab  = (const float*)d_in[25];
    const int* q_nb      = (const int*)d_in[26];
    const int* s_nb      = (const int*)d_in[27];
    const int* u_nb      = (const int*)d_in[28];
    const int* q_nb2     = (const int*)d_in[29];
    const int* user      = (const int*)d_in[30];
    const int* question  = (const int*)d_in[31];
    const int* response  = (const int*)d_in[32];
    const int* mask      = (const int*)d_in[33];
    float* out = (float*)d_out;

    // workspace carve-out (256B aligned)
    char* ws = (char*)d_ws;
    auto alloc = [&](size_t bytes) { char* p = ws; ws += (bytes + 255) & ~size_t(255); return p; };
    _Float16* h16    = (_Float16*)alloc(sizeof(_Float16) * Bv * PADv * KPv);
    _Float16* eq16   = (_Float16*)alloc(sizeof(_Float16) * Bv * PADv * KPv);
    _Float16* embs16 = (_Float16*)alloc(sizeof(_Float16) * PADv * KPv);
    _Float16* W16    = (_Float16*)alloc(sizeof(_Float16) * 4 * PADv * KPv);
    float* Sg   = (float*)alloc(sizeof(float) * Bv * PADv * PADv);
    float* Q0g  = (float*)alloc(sizeof(float) * Bv * PADv * PADv);
    float* lkv  = (float*)alloc(sizeof(float) * Bv * PADv);
    float* lq0  = (float*)alloc(sizeof(float) * Bv * PADv);
    float* lqs  = (float*)alloc(sizeof(float) * PADv);
    float* qwa  = (float*)alloc(sizeof(float) * 128);
    float* kwa  = (float*)alloc(sizeof(float) * 128);
    float* cst  = (float*)alloc(sizeof(float) * 2);
    unsigned* qbits = (unsigned*)alloc(sizeof(unsigned) * NQv * 4);

    // 1. zero the padded f16 operand buffers (h16 + eq16 contiguous, 16B granular)
    long n16 = (2L * Bv * PADv * KPv * (long)sizeof(_Float16)) / 16;
    zero16_kernel<<<256, 256, 0, stream>>>((uint4*)h16, n16);

    // 2. constants (attention rank-1 projections + emb_s f16 pad) & f16 weight pack
    const_prep_kernel<<<1, 256, 0, stream>>>(emb_s, qW, qb, kW, kb, wW,
                                             qwa, kwa, cst, lqs, embs16);
    wprep_kernel<<<(4 * PADv * KPv + 255) / 256, 256, 0, stream>>>(aggW, aggLW, W16);

    // 3. skill bitmasks
    qbits_kernel<<<(NQv + 255) / 256, 256, 0, stream>>>(qs_tab, qbits);

    // 4. phase 1: GNN (WMMA linears) + LSTM(h0=0) -> h rows (parallel over B x T)
    phase1_kernel<<<dim3(Tv, Bv), 128, 0, stream>>>(
        emb_q, emb_q2, emb_s, emb_u, emb_r,
        w1p, w2p, wcp, wpp, wnp,
        Wih, bih, bhh,
        W16, aggb, aggLb,
        uq_tab, q_nb, s_nb, u_nb, q_nb2,
        user, question, response, mask,
        qwa, kwa, cst, h16, eq16, lkv, lq0);

    // 5. batched WMMA GEMMs with fused sigmoid
    wmma_sig_kernel<<<dim3(PADv / 16, PADv / 16, Bv), 32, 0, stream>>>(embs16, 0L, h16, Sg);
    wmma_sig_kernel<<<dim3(PADv / 16, PADv / 16, Bv), 32, 0, stream>>>(eq16, (long)(PADv * KPv), h16, Q0g);

    // 6. rank-1 masked-softmax attention readout
    finalize_kernel<<<dim3(Tv, Bv), 128, 0, stream>>>(question, qbits, lkv, lq0, lqs, Sg, Q0g, out);
}
